// SpikeNetwork_56736517980954
// MI455X (gfx1250) — matrix operand order repaired and checked by reading, not compiled, not run on hardware
//
#include <hip/hip_runtime.h>

// ---------------------------------------------------------------------------
// SNN with Hebbian fast weights, MI455X (gfx1250, wave32, WMMA).
// - fastweight_kernel: roofline-dominant (32x768x768 f32 = 75.5MB, fits L2);
//   fused decay+rank1-update+readout, b128 vectorized, one read+write pass.
// - Big GEMMs ([32,768]x[768,768]) use v_wmma_f32_16x16x32_f16 with A in f16
//   (produced once per step) and W_h pre-packed into the exact B-fragment
//   layout -> one 32B coalesced load per lane per k-chunk.
// ---------------------------------------------------------------------------

typedef __attribute__((ext_vector_type(16))) _Float16 v16h;
typedef __attribute__((ext_vector_type(8)))  _Float16 v8h;
typedef __attribute__((ext_vector_type(8)))  float    v8f;
typedef __attribute__((ext_vector_type(4)))  float    v4f;

#define B_   32
#define NI_  512
#define NS_  50
#define NZ_  100
#define NH_  768
#define NO_  64
#define T_   24

#define THRESH_ 0.2f
#define VKEEP_  0.75f   // 1 - DT*VDECAY
#define REFRAC_ 2.0f

// ---------------------------- utility kernels ------------------------------

__global__ void zero_kernel(float* __restrict__ p, size_t n) {
  size_t i = (size_t)blockIdx.x * blockDim.x + threadIdx.x;
  size_t stride = (size_t)gridDim.x * blockDim.x;
  for (; i < n; i += stride) p[i] = 0.0f;
}

// Pack W_h[768,768] f32 into WMMA B-fragment order:
// Wpk[(((tn*24)+kc)*32+lane)*16 + 2j+e] = W_h[(kc*32 + 16*(lane>>4) + 2j+e)*768
//                                             + tn*16 + (lane&15)]
__global__ void pack_wh_kernel(const float* __restrict__ Wh,
                               _Float16* __restrict__ Wpk) {
  int t = blockIdx.x * blockDim.x + threadIdx.x;   // 589824 total
  if (t >= NH_ * NH_) return;
  int tn   = t / 12288;          // 24*32*16
  int r    = t % 12288;
  int kc   = r / 512;
  int r2   = r % 512;
  int lane = r2 / 16;
  int idx  = r2 % 16;
  int half = lane >> 4, lr = lane & 15;
  int kw = kc * 32 + 16 * half + idx;              // idx = 2j+e
  int n  = tn * 16 + lr;
  Wpk[t] = (_Float16)Wh[(size_t)kw * NH_ + n];
}

// ------------------------- small spiking layers ----------------------------
__global__ __launch_bounds__(256) void small_kernel(
    const float* __restrict__ xt, const float* __restrict__ Wi,
    const float* __restrict__ Wz,
    float* __restrict__ v_s, float* __restrict__ r_s, float* __restrict__ s,
    float* __restrict__ v_z, float* __restrict__ r_z, float* __restrict__ z) {
  int tid = threadIdx.x;
  for (int idx = tid; idx < B_ * NS_; idx += 256) {
    int b = idx / NS_, n = idx % NS_;
    float acc = 0.0f;
    for (int k = 0; k < NI_; ++k) acc += xt[b * NI_ + k] * Wi[k * NS_ + n];
    float r = fmaxf(r_s[idx] - 1.0f, 0.0f);
    float v = VKEEP_ * v_s[idx];
    if (r <= 0.0f) v += acc;
    float sp = (v >= THRESH_) ? 1.0f : 0.0f;
    if (sp > 0.0f) { r = REFRAC_; v = 0.0f; }
    v_s[idx] = v; r_s[idx] = r; s[idx] = sp;
  }
  __syncthreads();
  for (int idx = tid; idx < B_ * NZ_; idx += 256) {
    int b = idx / NZ_, n = idx % NZ_;
    float acc = 0.0f;
    for (int k = 0; k < NS_; ++k) acc += s[b * NS_ + k] * Wz[k * NZ_ + n];
    float r = fmaxf(r_z[idx] - 1.0f, 0.0f);
    float v = VKEEP_ * v_z[idx];
    if (r <= 0.0f) v += acc;
    float sp = (v >= THRESH_) ? 1.0f : 0.0f;
    if (sp > 0.0f) { r = REFRAC_; v = 0.0f; }
    v_z[idx] = v; r_z[idx] = r; z[idx] = sp;
  }
}

// zc = z @ W_c  [32,768], K=100.
__global__ __launch_bounds__(256) void zc_kernel(
    const float* __restrict__ z, const float* __restrict__ Wc,
    float* __restrict__ zc) {
  int idx = blockIdx.x * 256 + threadIdx.x;   // 96 blocks -> 24576
  int b = idx / NH_, n = idx % NH_;
  float acc = 0.0f;
  for (int k = 0; k < NZ_; ++k) acc += z[b * NZ_ + k] * Wc[k * NH_ + n];
  zc[idx] = acc;
}

// ----------------------------- WMMA GEMM -----------------------------------
// out = X16[32,768](f16) @ Wpk(B-fragment-packed W_h) + base[32,768]
// One wave per 16x16 tile; grid = 2 M-tiles * 48 N-tiles.
// mode 0: fused LIF_h epilogue -> spk, spk16, v_h, r_h.   mode 1: out.
__global__ __launch_bounds__(32) void gemm768_kernel(
    const _Float16* __restrict__ X16, const _Float16* __restrict__ Wpk,
    const float* __restrict__ base, float* __restrict__ out,
    float* __restrict__ v_h, float* __restrict__ r_h, float* __restrict__ spk,
    _Float16* __restrict__ spk16, int mode) {
  const int lane = threadIdx.x;
  const int lr   = lane & 15;
  const int half = lane >> 4;
  const int tn   = blockIdx.x % 48;
  const int tm   = blockIdx.x / 48;
  const int rowA = tm * 16 + lr;
  const int colB = tn * 16 + lr;

  const v16h* Wfrag = (const v16h*)Wpk + (size_t)tn * 24 * 32 + lane;
  const _Float16* xr = X16 + rowA * NH_ + 16 * half;

  v8f c = {0.f, 0.f, 0.f, 0.f, 0.f, 0.f, 0.f, 0.f};
  for (int kc = 0; kc < 24; ++kc) {
    // A 16x32 fragment in this code's contiguous k order (matches Wpk): elems 0-7 at K = kb+16*half+{0..7},
    //                          elems 8-15 at K = kb+16*half+8+{0..7}
    v8h alo = *(const v8h*)(xr + kc * 32);
    v8h ahi = *(const v8h*)(xr + kc * 32 + 8);
    v16h a = __builtin_shufflevector(alo, ahi, 0, 1, 2, 3, 4, 5, 6, 7,
                                     8, 9, 10, 11, 12, 13, 14, 15);
    v16h b = Wfrag[kc * 32];                 // one coalesced 32B load
    c = __builtin_amdgcn_wmma_f32_16x16x32_f16(
        /*neg_a=*/false, a, /*neg_b=*/false, b,
        /*c_mod=*/(short)0, c, /*reuse_a=*/false, /*reuse_b=*/false);
  }
  #pragma unroll
  for (int vv = 0; vv < 8; ++vv) {
    int m = tm * 16 + vv + 8 * half;         // C/D: M = vv + 8*half, N = lr
    int idx = m * NH_ + colB;
    float val = c[vv] + base[idx];
    if (mode == 0) {
      float r = fmaxf(r_h[idx] - 1.0f, 0.0f);
      float v = VKEEP_ * v_h[idx];
      if (r <= 0.0f) v += val;
      float sp = (v >= THRESH_) ? 1.0f : 0.0f;
      if (sp > 0.0f) { r = REFRAC_; v = 0.0f; }
      v_h[idx] = v; r_h[idx] = r; spk[idx] = sp;
      spk16[idx] = (_Float16)sp;
    } else {
      out[idx] = val;
    }
  }
}

// ----------------------- fused fast-weight streaming -----------------------
// a[b] = lam*a[b] + eta*outer(spk[b],spk[b]);  fw[b,:] += spk[b,:] @ a_new[b]
// Grid: 32 batches * 8 row-chunks (96 rows).  b128 loads/stores over j.
__global__ __launch_bounds__(192) void fastweight_kernel(
    const float* __restrict__ spk, float* __restrict__ a,
    float* __restrict__ fw, const float* __restrict__ lamp,
    const float* __restrict__ etap) {
  __shared__ float srow[NH_];
  int b = blockIdx.x >> 3;
  int chunk = blockIdx.x & 7;
  int tid = threadIdx.x;
  for (int j = tid; j < NH_; j += 192) srow[j] = spk[b * NH_ + j];
  __syncthreads();
  float lam = lamp[0], eta = etap[0];
  v4f sj = *(const v4f*)(srow + 4 * tid);
  v4f acc = {0.f, 0.f, 0.f, 0.f};
  float* abase = a + ((size_t)b * NH_ + (size_t)chunk * 96) * NH_ + 4 * tid;
  for (int i = 0; i < 96; ++i) {
    float si = srow[chunk * 96 + i];       // block-uniform -> scalar branch
    v4f* ap = (v4f*)(abase + (size_t)i * NH_);
    v4f av = *ap;
    if (si > 0.0f) {
      v4f nv = lam * av + (eta * si) * sj;
      *ap = nv;
      acc += si * nv;
    } else {
      *ap = lam * av;
    }
  }
  float* fwp = fw + b * NH_ + 4 * tid;
  atomicAdd(fwp + 0, acc[0]);
  atomicAdd(fwp + 1, acc[1]);
  atomicAdd(fwp + 2, acc[2]);
  atomicAdd(fwp + 3, acc[3]);
}

// ------------- batch-norm (over B) + relu + carry (f32 and f16) ------------
__global__ __launch_bounds__(NH_) void norm_kernel(
    const float* __restrict__ hs, const float* __restrict__ g,
    const float* __restrict__ bb, float* __restrict__ h,
    _Float16* __restrict__ h16) {
  int j = threadIdx.x;
  float vals[B_];
  float sum = 0.0f;
  #pragma unroll
  for (int b = 0; b < B_; ++b) { vals[b] = hs[b * NH_ + j]; sum += vals[b]; }
  float mu = sum * (1.0f / B_);
  float var = 0.0f;
  #pragma unroll
  for (int b = 0; b < B_; ++b) { float d = vals[b] - mu; var += d * d; }
  float sig = sqrtf(var * (1.0f / B_));
  float gj = g[j], bj = bb[j];
  #pragma unroll
  for (int b = 0; b < B_; ++b) {
    float y = fmaxf(gj * (vals[b] - mu) / sig + bj, 0.0f);
    h[b * NH_ + j] = y;
    h16[b * NH_ + j] = (_Float16)y;
  }
}

// ---------------------------- final readout --------------------------------
// LIF with zero state reduces to spike(x @ W >= THRESH).
__global__ __launch_bounds__(256) void final_kernel(
    const float* __restrict__ h, const float* __restrict__ Who,
    const float* __restrict__ Wo, float* __restrict__ o,
    float* __restrict__ y) {
  int tid = threadIdx.x;
  for (int idx = tid; idx < B_ * NZ_; idx += 256) {
    int b = idx / NZ_, n = idx % NZ_;
    float acc = 0.0f;
    for (int k = 0; k < NH_; ++k) acc += h[b * NH_ + k] * Who[k * NZ_ + n];
    o[idx] = (acc >= THRESH_) ? 1.0f : 0.0f;
  }
  __syncthreads();
  for (int idx = tid; idx < B_ * NO_; idx += 256) {
    int b = idx / NO_, n = idx % NO_;
    float acc = 0.0f;
    for (int k = 0; k < NZ_; ++k) acc += o[b * NZ_ + k] * Wo[k * NO_ + n];
    y[idx] = (acc >= THRESH_) ? 1.0f : 0.0f;
  }
}

// ------------------------------- launcher ----------------------------------

extern "C" void kernel_launch(void* const* d_in, const int* in_sizes, int n_in,
                              void* d_out, int out_size, void* d_ws,
                              size_t ws_size, hipStream_t stream) {
  const float* x_in = (const float*)d_in[0];
  const float* W_i  = (const float*)d_in[1];
  const float* W_z  = (const float*)d_in[2];
  const float* W_c  = (const float*)d_in[3];
  const float* W_h  = (const float*)d_in[4];
  const float* W_ho = (const float*)d_in[5];
  const float* W_o  = (const float*)d_in[6];
  const float* lam  = (const float*)d_in[7];
  const float* eta  = (const float*)d_in[8];
  const float* g    = (const float*)d_in[9];
  const float* bb   = (const float*)d_in[10];
  // d_in[11] = time = 24 (compile-time constant per reference)

  float* W = (float*)d_ws;
  const size_t A_N = (size_t)B_ * NH_ * NH_;            // 18,874,368 floats
  // all sub-buffer sizes are multiples of 4 floats -> 16B alignment holds
  float* a    = W;
  float* v_s  = W + A_N;
  float* r_s  = v_s + B_ * NS_;
  float* v_z  = r_s + B_ * NS_;
  float* r_z  = v_z + B_ * NZ_;
  float* v_h  = r_z + B_ * NZ_;
  float* r_h  = v_h + B_ * NH_;
  float* h    = r_h + B_ * NH_;
  _Float16* h16 = (_Float16*)(h + B_ * NH_);            // 24576 halves
  const size_t ZERO_N = (size_t)((float*)(h16 + B_ * NH_) - W);
  float* s    = W + ZERO_N;
  float* z    = s + B_ * NS_;
  float* zc   = z + B_ * NZ_;
  float* spk  = zc + B_ * NH_;
  _Float16* spk16 = (_Float16*)(spk + B_ * NH_);        // 24576 halves
  float* fw   = (float*)(spk16 + B_ * NH_);
  float* o    = fw + B_ * NH_;
  _Float16* Wpk = (_Float16*)(o + B_ * NZ_);            // 589824 halves

  // deterministic init: zero fast weights + all carried state (incl. h16)
  zero_kernel<<<4096, 256, 0, stream>>>(W, ZERO_N);
  pack_wh_kernel<<<(NH_ * NH_ + 255) / 256, 256, 0, stream>>>(W_h, Wpk);

  for (int t = 0; t < T_; ++t) {
    const float* xt = x_in + (size_t)t * B_ * NI_;
    small_kernel<<<1, 256, 0, stream>>>(xt, W_i, W_z, v_s, r_s, s, v_z, r_z, z);
    zc_kernel<<<(B_ * NH_) / 256, 256, 0, stream>>>(z, W_c, zc);
    // u = h_prev @ W_h + zc -> LIF -> spk (+f16 copy)
    gemm768_kernel<<<96, 32, 0, stream>>>(h16, Wpk, zc, fw, v_h, r_h, spk,
                                          spk16, 0);
    // fw = spk @ W_h + zc
    gemm768_kernel<<<96, 32, 0, stream>>>(spk16, Wpk, zc, fw, v_h, r_h, spk,
                                          spk16, 1);
    // a = lam*a + eta*outer(spk,spk);  fw += spk @ a_new
    fastweight_kernel<<<256, 192, 0, stream>>>(spk, a, fw, lam, eta);
    // batch-norm over B + relu -> new h carry (f32 + f16)
    norm_kernel<<<1, NH_, 0, stream>>>(fw, g, bb, h, h16);
  }

  final_kernel<<<1, 256, 0, stream>>>(h, W_ho, W_o, o, (float*)d_out);
}
